// MeshLaplacianLoss_8521215115747
// MI455X (gfx1250) — matrix operand order repaired and checked by reading, not compiled, not run on hardware
//
#include <hip/hip_runtime.h>
#include <hip/hip_bf16.h>

// ---- CDNA5 WMMA vector types (wave32) ----
typedef __attribute__((ext_vector_type(16))) __bf16 v16bf;  // A/B operand: 16x32 / 32x16 bf16
typedef __attribute__((ext_vector_type(8)))  float  v8f;    // C/D: 16x16 f32

#define NVERT   9216
#define NCOL    48          // 16 batches x 3 coords
#define NBATCH  16
#define KBLK    32          // K per bf16 WMMA
#define WAVES   8           // waves per workgroup (K-split)
#define KSLAB   (NVERT / WAVES)      // 1152
#define NSTEPS  (KSLAB / KBLK)       // 36
#define NBLOCKS (NVERT / 16)         // 576 row-blocks
#define PACKED_ELEMS ((size_t)NVERT * NCOL)  // 442368 per (hi|lo) array

// -------------------------------------------------------------------------
// Pre-pass: split verts (f32) into bf16 hi + lo and scatter into the exact
// per-lane VGPR layout the 16x16x32 bf16 WMMA B-operand expects, so the main
// loop's B load is one contiguous 32B read per lane.
//   B tile (32x16): lane = (Krow&8 ? 16:0) + col%16 ; element i = (Krow&7) + (Krow>=16 ? 8:0)
// Also zeroes the output scalar (d_out is poisoned by the harness).
// -------------------------------------------------------------------------
__global__ void pack_verts_kernel(const float* __restrict__ verts,
                                  __bf16* __restrict__ xhi,
                                  __bf16* __restrict__ xlo,
                                  float* __restrict__ out) {
  int id = blockIdx.x * blockDim.x + threadIdx.x;
  if (id == 0) out[0] = 0.0f;
  if (id >= (int)(NVERT * NCOL)) return;
  int w   = id / NCOL;          // K index (vertex)
  int col = id - w * NCOL;      // N index = batch*3 + coord
  int b = col / 3;
  int c = col - b * 3;
  float v = verts[((size_t)b * NVERT + w) * 3 + c];
  __bf16 hi = (__bf16)v;
  __bf16 lo = (__bf16)(v - (float)hi);
  int kblk = w >> 5;            // which 32-wide K block
  int r    = w & 31;            // K row within block
  int j    = col >> 4;          // N tile (0..2)
  int n    = col & 15;          // column within tile
  int lane = ((r & 8) << 1) + n; // lanes 0-15: K {0..7,16..23}, lanes 16-31: K {8..15,24..31}
  int i    = (r & 7) | ((r >> 1) & 8); // element index within lane's 16 bf16
  size_t idx = ((size_t)(kblk * 3 + j) * 32 + lane) * 16 + i;
  xhi[idx] = hi;
  xlo[idx] = lo;
}

// -------------------------------------------------------------------------
// Main: per 16-row block of L, 8 waves K-split; bf16 WMMA with f32 accum.
// acc += A_bf16 x Bhi  +  A_bf16 x Blo   (exact: L entries are small ints)
// -------------------------------------------------------------------------
__global__ void __launch_bounds__(256, 2)
lap_gemm_kernel(const float* __restrict__ L,
                const __bf16* __restrict__ xhi,
                const __bf16* __restrict__ xlo,
                float* __restrict__ partials) {
  __shared__ float red[WAVES][3 * 8 * 32];   // 24.6 KB: per-wave partial C tiles
  __shared__ float wsum[WAVES];

  const int tid  = threadIdx.x;
  const int wave = tid >> 5;
  const int lane = tid & 31;

  // A operand (16x32 bf16) per-lane addressing per ISA layout:
  // lanes 0-15:  rows 0..15, K = {0..7, 16..23}
  // lanes 16-31: rows 0..15, K = {8..15, 24..31}
  const int m0   = blockIdx.x * 16;
  const int row  = m0 + (lane & 15);
  const int cofs = (lane >> 4) * 8;
  const size_t rowbase = (size_t)row * NVERT;

  const v16bf* __restrict__ bph = (const v16bf*)xhi;
  const v16bf* __restrict__ bpl = (const v16bf*)xlo;

  v8f acc0 = {}, acc1 = {}, acc2 = {};

  const int k0 = wave * KSLAB;
  for (int s = 0; s < NSTEPS; ++s) {
    const int k = k0 + s * KBLK;
    const float* lp = L + rowbase + k + cofs;
    __builtin_prefetch(lp + KBLK, 0, 0);   // global_prefetch_b8: next A tile

    float4 f0 = *(const float4*)(lp + 0);
    float4 f1 = *(const float4*)(lp + 4);
    float4 f2 = *(const float4*)(lp + 16);
    float4 f3 = *(const float4*)(lp + 20);

    v16bf a;   // bf16(L) is exact: entries are {-1, 0, degree<=8}
    a[0]  = (__bf16)f0.x; a[1]  = (__bf16)f0.y; a[2]  = (__bf16)f0.z; a[3]  = (__bf16)f0.w;
    a[4]  = (__bf16)f1.x; a[5]  = (__bf16)f1.y; a[6]  = (__bf16)f1.z; a[7]  = (__bf16)f1.w;
    a[8]  = (__bf16)f2.x; a[9]  = (__bf16)f2.y; a[10] = (__bf16)f2.z; a[11] = (__bf16)f2.w;
    a[12] = (__bf16)f3.x; a[13] = (__bf16)f3.y; a[14] = (__bf16)f3.z; a[15] = (__bf16)f3.w;

    const size_t bbase = (size_t)((k >> 5) * 3) * 32 + lane;
    v16bf bh0 = bph[bbase];      v16bf bl0 = bpl[bbase];
    v16bf bh1 = bph[bbase + 32]; v16bf bl1 = bpl[bbase + 32];
    v16bf bh2 = bph[bbase + 64]; v16bf bl2 = bpl[bbase + 64];

    acc0 = __builtin_amdgcn_wmma_f32_16x16x32_bf16(false, a, false, bh0, (short)0, acc0, false, false);
    acc0 = __builtin_amdgcn_wmma_f32_16x16x32_bf16(false, a, false, bl0, (short)0, acc0, false, false);
    acc1 = __builtin_amdgcn_wmma_f32_16x16x32_bf16(false, a, false, bh1, (short)0, acc1, false, false);
    acc1 = __builtin_amdgcn_wmma_f32_16x16x32_bf16(false, a, false, bl1, (short)0, acc1, false, false);
    acc2 = __builtin_amdgcn_wmma_f32_16x16x32_bf16(false, a, false, bh2, (short)0, acc2, false, false);
    acc2 = __builtin_amdgcn_wmma_f32_16x16x32_bf16(false, a, false, bl2, (short)0, acc2, false, false);
  }

  // Cross-wave K reduction in LDS (layout-agnostic: all waves share layout),
  // then sum of squares for this 16x48 block.
#pragma unroll
  for (int i = 0; i < 8; ++i) {
    red[wave][(0 * 8 + i) * 32 + lane] = acc0[i];
    red[wave][(1 * 8 + i) * 32 + lane] = acc1[i];
    red[wave][(2 * 8 + i) * 32 + lane] = acc2[i];
  }
  __syncthreads();

  float s = 0.0f;
  for (int e = tid; e < 3 * 8 * 32; e += 256) {
    float v = 0.0f;
#pragma unroll
    for (int w2 = 0; w2 < WAVES; ++w2) v += red[w2][e];
    s += v * v;
  }
#pragma unroll
  for (int off = 16; off > 0; off >>= 1) s += __shfl_down(s, off, 32);
  if (lane == 0) wsum[wave] = s;
  __syncthreads();
  if (tid == 0) {
    float t = 0.0f;
#pragma unroll
    for (int w2 = 0; w2 < WAVES; ++w2) t += wsum[w2];
    partials[blockIdx.x] = t;
  }
}

// -------------------------------------------------------------------------
// Deterministic final reduction of 576 block partials.
// -------------------------------------------------------------------------
__global__ void final_reduce_kernel(const float* __restrict__ partials,
                                    float* __restrict__ out) {
  __shared__ float sm[256];
  float s = 0.0f;
  for (int i = threadIdx.x; i < NBLOCKS; i += 256) s += partials[i];
  sm[threadIdx.x] = s;
  __syncthreads();
  for (int off = 128; off > 0; off >>= 1) {
    if (threadIdx.x < off) sm[threadIdx.x] += sm[threadIdx.x + off];
    __syncthreads();
  }
  if (threadIdx.x == 0) out[0] = sm[0] * (1.0f / (float)NBATCH);
}

extern "C" void kernel_launch(void* const* d_in, const int* in_sizes, int n_in,
                              void* d_out, int out_size, void* d_ws, size_t ws_size,
                              hipStream_t stream) {
  const float* verts = (const float*)d_in[0];   // [16, 9216, 3] f32
  const float* L     = (const float*)d_in[1];   // [9216, 9216] f32
  float* out         = (float*)d_out;           // scalar f32

  __bf16* xhi     = (__bf16*)d_ws;
  __bf16* xlo     = xhi + PACKED_ELEMS;
  float*  partials = (float*)(xlo + PACKED_ELEMS);  // 2*442368*2B is 4B-aligned

  int packThreads = (int)(NVERT * NCOL);
  pack_verts_kernel<<<(packThreads + 255) / 256, 256, 0, stream>>>(verts, xhi, xlo, out);
  lap_gemm_kernel<<<NBLOCKS, 256, 0, stream>>>(L, xhi, xlo, partials);
  final_reduce_kernel<<<1, 256, 0, stream>>>(partials, out);
}
